// Ouroboros6Model_68685116998085
// MI455X (gfx1250) — compile-verified, matrix operand order checked by reading, not run on hardware
//
#include <hip/hip_runtime.h>
#include <hip/hip_bf16.h>

#define Bv   2
#define Tv   1024
#define Dv   1024
#define Hv   16
#define HDv  64
#define Lv   6
#define HIDv 2816
#define Kv   512            // top-k tokens per batch
#define EPSv 1e-5f
#define LOG_THETA 9.210340371976184f   // ln(10000)

typedef __attribute__((ext_vector_type(16))) __bf16 v16bf;
typedef __attribute__((ext_vector_type(8)))  __bf16 v8bf;
typedef __attribute__((ext_vector_type(4)))  __bf16 v4bf;
typedef __attribute__((ext_vector_type(8)))  float  v8f;

// ---------------------------------------------------------------------------
// WMMA helpers (CDNA5: D = A(16x32 bf16) * B(32x16 bf16) + C(16x16 f32))
// ---------------------------------------------------------------------------
__device__ __forceinline__ v8f wmma_bf16(v16bf a, v16bf b, v8f c) {
    return __builtin_amdgcn_wmma_f32_16x16x32_bf16(
        /*neg_a=*/false, a, /*neg_b=*/false, b,
        /*c_mod=*/(short)0, c, /*reuse_a=*/false, /*reuse_b=*/false);
}

// Load a 16x32 fragment from a bf16 row (K-contiguous).
// Per ISA layout: lane m = lane%16; halves 0..7 <- K = hw*8 + 0..7,
// halves 8..15 <- K = 16 + hw*8 + 0..7.  p points at row + k0 + hw*8.
__device__ __forceinline__ v16bf frag_from_bf16(const __bf16* __restrict__ p) {
    v8bf lo = *(const v8bf*)(p);
    v8bf hi = *(const v8bf*)(p + 16);
    return __builtin_shufflevector(lo, hi, 0,1,2,3,4,5,6,7,8,9,10,11,12,13,14,15);
}

// ---------------------------------------------------------------------------
// GEMM:  C[M,N] (+)= A[M,K] * B[N,K]^T   (bf16 operands, f32 accumulate)
// block = 128 threads = 4 waves (2x2), each wave computes a 64x64 tile
// grid  = (N/128, M/128); M%128==0, N%128==0, K%32==0 by construction
// ---------------------------------------------------------------------------
__global__ __launch_bounds__(128) void gemm_nt_kernel(
    const __bf16* __restrict__ A, const __bf16* __restrict__ Bw,
    float* __restrict__ C, int M, int N, int K, int acc)
{
    const int tid  = threadIdx.x;
    const int lane = tid & 31;
    const int wv   = tid >> 5;
    const int n16  = lane & 15;
    const int hw   = lane >> 4;
    const int m0 = blockIdx.y * 128 + (wv >> 1) * 64;
    const int n0 = blockIdx.x * 128 + (wv &  1) * 64;

    v8f c[4][4];
#pragma unroll
    for (int i = 0; i < 4; ++i)
#pragma unroll
        for (int j = 0; j < 4; ++j) c[i][j] = (v8f)0.0f;

    const __bf16* arow[4];
    const __bf16* brow[4];
#pragma unroll
    for (int i = 0; i < 4; ++i)
        arow[i] = A + (size_t)(m0 + i*16 + n16) * K + hw * 8;
#pragma unroll
    for (int j = 0; j < 4; ++j)
        brow[j] = Bw + (size_t)(n0 + j*16 + n16) * K + hw * 8;

    for (int k0 = 0; k0 < K; k0 += 32) {
        v16bf a[4], b[4];
#pragma unroll
        for (int i = 0; i < 4; ++i) a[i] = frag_from_bf16(arow[i] + k0);
#pragma unroll
        for (int j = 0; j < 4; ++j) b[j] = frag_from_bf16(brow[j] + k0);
        if (k0 + 32 < K) {   // gfx1250 global_prefetch_b8 for next K step
#pragma unroll
            for (int i = 0; i < 4; ++i) __builtin_prefetch(arow[i] + k0 + 32, 0, 1);
#pragma unroll
            for (int j = 0; j < 4; ++j) __builtin_prefetch(brow[j] + k0 + 32, 0, 1);
        }
#pragma unroll
        for (int i = 0; i < 4; ++i)
#pragma unroll
            for (int j = 0; j < 4; ++j)
                c[i][j] = wmma_bf16(a[i], b[j], c[i][j]);
    }

#pragma unroll
    for (int i = 0; i < 4; ++i) {
        const int mb = m0 + i*16 + hw*8;
#pragma unroll
        for (int j = 0; j < 4; ++j) {
            const int n = n0 + j*16 + n16;
#pragma unroll
            for (int e = 0; e < 8; ++e) {
                float* dst = C + (size_t)(mb + e) * N + n;
                if (acc) *dst += c[i][j][e]; else *dst = c[i][j][e];
            }
        }
    }
}

// ---------------------------------------------------------------------------
// Flash attention (causal). qkv_bf: [B,T,3*D] bf16 with rope applied.
// One wave per (b,h,query-tile of 16). 8 waves / block. Output bf16.
// ---------------------------------------------------------------------------
__global__ __launch_bounds__(256) void attn_kernel(
    const __bf16* __restrict__ qkv, __bf16* __restrict__ O)
{
    __shared__ __bf16 Pbuf[8][16 * 32];   // per-wave P staging (C-layout -> A-frag)
    __shared__ __bf16 Vbuf[8][32 * 64];   // per-wave V tile (keys x head-dim)

    const int tid  = threadIdx.x;
    const int lane = tid & 31;
    const int wv   = tid >> 5;
    const int n16  = lane & 15;
    const int hw   = lane >> 4;

    const int gw = blockIdx.x * 8 + wv;
    const int qt = gw & 63;            // T/16 = 64 query tiles
    const int bh = gw >> 6;
    const int b  = bh >> 4;            // H = 16
    const int h  = bh & 15;
    const int q0 = qt * 16;

    const __bf16* base = qkv + (size_t)b * Tv * 3 * Dv;
    __bf16* Pl = Pbuf[wv];
    __bf16* Vl = Vbuf[wv];

    v16bf qf0, qf1;
    {
        const __bf16* qrow = base + (size_t)(q0 + n16) * 3 * Dv + h * HDv;
        qf0 = frag_from_bf16(qrow + hw * 8);
        qf1 = frag_from_bf16(qrow + 32 + hw * 8);
    }

    v8f o[4];
#pragma unroll
    for (int j = 0; j < 4; ++j) o[j] = (v8f)0.0f;
    float mrow[8], lrow[8];
#pragma unroll
    for (int i = 0; i < 8; ++i) { mrow[i] = -__builtin_inff(); lrow[i] = 0.0f; }

    const int nkb = (q0 + 16 + 31) >> 5;      // key blocks of 32 (causal)
    for (int kb = 0; kb < nkb; ++kb) {
        const int kb0 = kb * 32;

        // S = Q * K^T  (two 16-key subtiles, K rows are d-contiguous)
        v8f S[2];
#pragma unroll
        for (int s = 0; s < 2; ++s) {
            const __bf16* krow =
                base + (size_t)(kb0 + s*16 + n16) * 3 * Dv + Dv + h * HDv;
            v16bf kf0 = frag_from_bf16(krow + hw * 8);
            v16bf kf1 = frag_from_bf16(krow + 32 + hw * 8);
            v8f t = (v8f)0.0f;
            t = wmma_bf16(qf0, kf0, t);
            t = wmma_bf16(qf1, kf1, t);
            S[s] = t;
        }

        // stage V tile: each lane copies one key row (64 halves, b128 chunks)
        {
            const __bf16* vrow =
                base + (size_t)(kb0 + lane) * 3 * Dv + 2 * Dv + h * HDv;
#pragma unroll
            for (int c8 = 0; c8 < 8; ++c8)
                *(v8bf*)(Vl + lane * 64 + c8 * 8) = *(const v8bf*)(vrow + c8 * 8);
        }

        // scale + causal mask + online softmax (row = i + hw*8, col = kb0+s*16+n16)
        float alpha[8];
#pragma unroll
        for (int i = 0; i < 8; ++i) {
            const int qr = q0 + i + hw * 8;
            float s0 = S[0][i] * 0.125f;
            float s1 = S[1][i] * 0.125f;
            if (kb0 + n16      > qr) s0 = -1e30f;
            if (kb0 + 16 + n16 > qr) s1 = -1e30f;
            float mx = fmaxf(s0, s1);
            mx = fmaxf(mx, __shfl_xor(mx, 1));
            mx = fmaxf(mx, __shfl_xor(mx, 2));
            mx = fmaxf(mx, __shfl_xor(mx, 4));
            mx = fmaxf(mx, __shfl_xor(mx, 8));
            const float mnew = fmaxf(mrow[i], mx);
            alpha[i] = __expf(mrow[i] - mnew);
            const float p0 = __expf(s0 - mnew);
            const float p1 = __expf(s1 - mnew);
            float rs = p0 + p1;
            rs += __shfl_xor(rs, 1);
            rs += __shfl_xor(rs, 2);
            rs += __shfl_xor(rs, 4);
            rs += __shfl_xor(rs, 8);
            lrow[i] = lrow[i] * alpha[i] + rs;
            mrow[i] = mnew;
            S[0][i] = p0; S[1][i] = p1;
        }
#pragma unroll
        for (int j = 0; j < 4; ++j)
#pragma unroll
            for (int i = 0; i < 8; ++i) o[j][i] *= alpha[i];

        // stage P (C-layout -> row-major 16x32 bf16 in LDS)
#pragma unroll
        for (int i = 0; i < 8; ++i) {
            const int m = i + hw * 8;
            Pl[m * 32 + n16]      = (__bf16)S[0][i];
            Pl[m * 32 + 16 + n16] = (__bf16)S[1][i];
        }
        asm volatile("s_wait_dscnt 0" ::: "memory");

        // reload P as an A fragment
        v16bf pf;
        {
            const __bf16* pr = Pl + n16 * 32 + hw * 8;
            v8bf lo = *(const v8bf*)(pr);
            v8bf hi = *(const v8bf*)(pr + 16);
            pf = __builtin_shufflevector(lo, hi,
                 0,1,2,3,4,5,6,7,8,9,10,11,12,13,14,15);
        }

        // O += P * V : B fragments (32 keys x 16 d) via LDS transpose
#pragma unroll
        for (int j = 0; j < 4; ++j) {
            v16bf vf;
#pragma unroll
            for (int hh = 0; hh < 16; ++hh) {
                const int k = (hh < 8) ? (hw * 8 + hh) : (16 + hw * 8 + (hh - 8));
                vf[hh] = Vl[k * 64 + j * 16 + n16];
            }
            o[j] = wmma_bf16(pf, vf, o[j]);
        }
    }

    // write O / l  (bf16, consumed directly by the wo GEMM)
#pragma unroll
    for (int e = 0; e < 8; ++e) {
        const float inv_l = 1.0f / lrow[e];
        const size_t row = (size_t)b * Tv + q0 + e + hw * 8;
#pragma unroll
        for (int j = 0; j < 4; ++j)
            O[row * Dv + h * HDv + j * 16 + n16] = (__bf16)(o[j][e] * inv_l);
    }
}

// ---------------------------------------------------------------------------
// Small kernels
// ---------------------------------------------------------------------------
__global__ __launch_bounds__(256) void f32_to_bf16_kernel(
    const float* __restrict__ src, __bf16* __restrict__ dst, int n)
{
    const int i = (blockIdx.x * 256 + threadIdx.x) * 4;   // n % 4 == 0 always
    if (i < n) {
        float4 v = *(const float4*)(src + i);
        v4bf o;
        o[0] = (__bf16)v.x; o[1] = (__bf16)v.y;
        o[2] = (__bf16)v.z; o[3] = (__bf16)v.w;
        *(v4bf*)(dst + i) = o;
    }
}

__global__ __launch_bounds__(256) void embed_kernel(
    const int* __restrict__ ids, const int* __restrict__ iter,
    const float* __restrict__ emb, const float* __restrict__ itemb,
    float* __restrict__ x)
{
    const int r  = blockIdx.x;
    const int id = ids[r];
    const int it = iter[0];
    for (int d = threadIdx.x; d < Dv; d += 256)
        x[(size_t)r * Dv + d] = emb[(size_t)id * Dv + d] + itemb[(size_t)it * Dv + d];
}

// RMSNorm; if sel_idx != nullptr, gathers row (b = r/Kp, t = sel_idx[r]).
template <typename OutT>
__global__ __launch_bounds__(256) void rmsnorm_kernel(
    const float* __restrict__ x, const float* __restrict__ w,
    OutT* __restrict__ out, const int* __restrict__ sel_idx, int Kp)
{
    __shared__ float red[8];
    const int r = blockIdx.x;
    const float* src;
    if (sel_idx) {
        const int b = r / Kp;
        src = x + ((size_t)b * Tv + sel_idx[r]) * Dv;
    } else {
        src = x + (size_t)r * Dv;
    }
    float ss = 0.0f;
    for (int d = threadIdx.x; d < Dv; d += 256) { float v = src[d]; ss += v * v; }
    ss += __shfl_xor(ss, 1);  ss += __shfl_xor(ss, 2);
    ss += __shfl_xor(ss, 4);  ss += __shfl_xor(ss, 8);
    ss += __shfl_xor(ss, 16);
    if ((threadIdx.x & 31) == 0) red[threadIdx.x >> 5] = ss;
    __syncthreads();
    float tot = 0.0f;
#pragma unroll
    for (int i = 0; i < 8; ++i) tot += red[i];
    const float norm = rsqrtf(tot / (float)Dv + EPSv);
    for (int d = threadIdx.x; d < Dv; d += 256)
        out[(size_t)r * Dv + d] = (OutT)(src[d] * norm * w[d]);
}

// RoPE on q,k + cast qkv f32 -> bf16
__global__ __launch_bounds__(256) void rope_cast_kernel(
    const float* __restrict__ qkv, __bf16* __restrict__ out)
{
    const int r = blockIdx.x;
    const int t = r & (Tv - 1);
    const float* src = qkv + (size_t)r * 3 * Dv;
    __bf16* dst = out + (size_t)r * 3 * Dv;
    for (int p = threadIdx.x; p < 1024; p += 256) {   // 512 q-pairs + 512 k-pairs
        const int which = p >> 9;
        const int pr = p & 511;
        const int i  = pr & 31;            // pair index within head (HD/2=32)
        const int hh = pr >> 5;            // head
        const int e  = which * Dv + hh * HDv + 2 * i;
        const float ang = (float)t * __expf(-((2.0f * (float)i) / 64.0f) * LOG_THETA);
        float sn, cs;
        __sincosf(ang, &sn, &cs);
        const float xr = src[e], xi = src[e + 1];
        dst[e]     = (__bf16)(xr * cs - xi * sn);
        dst[e + 1] = (__bf16)(xr * sn + xi * cs);
    }
    for (int d = threadIdx.x; d < Dv; d += 256)
        dst[2 * Dv + d] = (__bf16)src[2 * Dv + d];
}

__global__ __launch_bounds__(256) void router_kernel(
    const float* __restrict__ x, const float* __restrict__ rw,
    float* __restrict__ probs)
{
    __shared__ float red[8];
    const int r = blockIdx.x;
    const float* src = x + (size_t)r * Dv;
    float s = 0.0f;
    for (int d = threadIdx.x; d < Dv; d += 256) s += src[d] * rw[d];
    s += __shfl_xor(s, 1);  s += __shfl_xor(s, 2);
    s += __shfl_xor(s, 4);  s += __shfl_xor(s, 8);
    s += __shfl_xor(s, 16);
    if ((threadIdx.x & 31) == 0) red[threadIdx.x >> 5] = s;
    __syncthreads();
    if (threadIdx.x == 0) {
        float tot = 0.0f;
#pragma unroll
        for (int i = 0; i < 8; ++i) tot += red[i];
        probs[r] = 1.0f / (1.0f + __expf(-tot));
    }
}

// Exact stable top-K by rank counting (matches lax.top_k tie-break)
__global__ __launch_bounds__(256) void topk_kernel(
    const float* __restrict__ probs, int* __restrict__ sel_idx,
    float* __restrict__ sel_prob)
{
    __shared__ float p[Tv];
    const int b = blockIdx.x;
    for (int t = threadIdx.x; t < Tv; t += 256) p[t] = probs[b * Tv + t];
    __syncthreads();
    for (int t = threadIdx.x; t < Tv; t += 256) {
        const float pt = p[t];
        int rank = 0;
        for (int j = 0; j < Tv; ++j)
            rank += (p[j] > pt) || (p[j] == pt && j < t);
        if (rank < Kv) {
            sel_idx[b * Kv + rank]  = t;
            sel_prob[b * Kv + rank] = pt;
        }
    }
}

__global__ __launch_bounds__(256) void silu_mul_kernel(
    const float* __restrict__ gate, const float* __restrict__ up,
    __bf16* __restrict__ gu, int n)
{
    const int i = blockIdx.x * 256 + threadIdx.x;
    if (i < n) {
        const float g = gate[i];
        gu[i] = (__bf16)((g / (1.0f + __expf(-g))) * up[i]);
    }
}

__global__ __launch_bounds__(256) void scatter_kernel(
    const float* __restrict__ mlp, const int* __restrict__ sel_idx,
    const float* __restrict__ sel_prob, float* __restrict__ x)
{
    const int r = blockIdx.x;            // b*Kv + k
    const int b = r / Kv;
    const int t = sel_idx[r];
    const float pscale = sel_prob[r];
    float* dst = x + ((size_t)b * Tv + t) * Dv;
    const float* src = mlp + (size_t)r * Dv;
    for (int d = threadIdx.x; d < Dv; d += 256)
        dst[d] += src[d] * pscale;
}

// ---------------------------------------------------------------------------
// Host launcher
// ---------------------------------------------------------------------------
extern "C" void kernel_launch(void* const* d_in, const int* in_sizes, int n_in,
                              void* d_out, int out_size, void* d_ws, size_t ws_size,
                              hipStream_t stream)
{
    (void)in_sizes; (void)n_in; (void)out_size; (void)ws_size;
    const int*   ids      = (const int*)  d_in[0];
    const int*   iter     = (const int*)  d_in[1];
    const float* emb      = (const float*)d_in[2];
    const float* itemb    = (const float*)d_in[3];
    const float* attn_nw  = (const float*)d_in[4];
    const float* Wqkv     = (const float*)d_in[5];
    const float* wo       = (const float*)d_in[6];
    const float* router_w = (const float*)d_in[7];
    const float* mlp_nw   = (const float*)d_in[8];
    const float* gate_w   = (const float*)d_in[9];
    const float* up_w     = (const float*)d_in[10];
    const float* down_w   = (const float*)d_in[11];
    const float* final_nw = (const float*)d_in[12];

    const int BT = Bv * Tv;       // 2048
    const int BK = Bv * Kv;       // 1024
    char* w = (char*)d_ws;
    size_t off = 0;
    auto alloc = [&](size_t bytes) { char* p = w + off; off += (bytes + 255) & ~(size_t)255; return p; };

    float*  x      = (float*) alloc((size_t)BT * Dv * 4);
    __bf16* hbf    = (__bf16*)alloc((size_t)BT * Dv * 2);
    float*  qkv    = (float*) alloc((size_t)BT * 3 * Dv * 4);
    __bf16* qkv_bf = (__bf16*)alloc((size_t)BT * 3 * Dv * 2);
    __bf16* obf    = (__bf16*)alloc((size_t)BT * Dv * 2);
    float*  probs  = (float*) alloc((size_t)BT * 4);
    int*    selidx = (int*)   alloc((size_t)BK * 4);
    float*  selpr  = (float*) alloc((size_t)BK * 4);
    __bf16* hmbf   = (__bf16*)alloc((size_t)BK * Dv * 2);
    float*  gate   = (float*) alloc((size_t)BK * HIDv * 4);
    float*  up     = (float*) alloc((size_t)BK * HIDv * 4);
    __bf16* gubf   = (__bf16*)alloc((size_t)BK * HIDv * 2);
    float*  mlp    = (float*) alloc((size_t)BK * Dv * 4);

    // per-layer bf16 weight staging (L2-resident: ~26 MB)
    const size_t nWqkv = (size_t)3 * Dv * Dv;     // 3.15M
    const size_t nWo   = (size_t)Dv * Dv;         // 1.05M
    const size_t nWg   = (size_t)HIDv * Dv;       // 2.88M (gate / up / down each)
    __bf16* wqkv_bf = (__bf16*)alloc(nWqkv * 2);
    __bf16* wo_bf   = (__bf16*)alloc(nWo * 2);
    __bf16* wg_bf   = (__bf16*)alloc(nWg * 2);
    __bf16* wu_bf   = (__bf16*)alloc(nWg * 2);
    __bf16* wd_bf   = (__bf16*)alloc(nWg * 2);

    auto cvt = [&](const float* s, __bf16* d, size_t n) {
        f32_to_bf16_kernel<<<(int)((n / 4 + 255) / 256), 256, 0, stream>>>(s, d, (int)n);
    };

    embed_kernel<<<BT, 256, 0, stream>>>(ids, iter, emb, itemb, x);

    for (int l = 0; l < Lv; ++l) {
        // stage this layer's weights as bf16 (once per call)
        cvt(Wqkv   + (size_t)l * nWqkv, wqkv_bf, nWqkv);
        cvt(wo     + (size_t)l * nWo,   wo_bf,   nWo);
        cvt(gate_w + (size_t)l * nWg,   wg_bf,   nWg);
        cvt(up_w   + (size_t)l * nWg,   wu_bf,   nWg);
        cvt(down_w + (size_t)l * nWg,   wd_bf,   nWg);

        rmsnorm_kernel<__bf16><<<BT, 256, 0, stream>>>(
            x, attn_nw + (size_t)l * Dv, hbf, nullptr, 0);

        gemm_nt_kernel<<<dim3(3 * Dv / 128, BT / 128), 128, 0, stream>>>(
            hbf, wqkv_bf, qkv, BT, 3 * Dv, Dv, 0);

        rope_cast_kernel<<<BT, 256, 0, stream>>>(qkv, qkv_bf);

        attn_kernel<<<Bv * Hv * (Tv / 16) / 8, 256, 0, stream>>>(qkv_bf, obf);

        gemm_nt_kernel<<<dim3(Dv / 128, BT / 128), 128, 0, stream>>>(
            obf, wo_bf, x, BT, Dv, Dv, 1);   // residual add

        router_kernel<<<BT, 256, 0, stream>>>(x, router_w + (size_t)l * Dv, probs);
        topk_kernel<<<Bv, 256, 0, stream>>>(probs, selidx, selpr);

        rmsnorm_kernel<__bf16><<<BK, 256, 0, stream>>>(
            x, mlp_nw + (size_t)l * Dv, hmbf, selidx, Kv);

        gemm_nt_kernel<<<dim3(HIDv / 128, BK / 128), 128, 0, stream>>>(
            hmbf, wg_bf, gate, BK, HIDv, Dv, 0);
        gemm_nt_kernel<<<dim3(HIDv / 128, BK / 128), 128, 0, stream>>>(
            hmbf, wu_bf, up, BK, HIDv, Dv, 0);

        silu_mul_kernel<<<(BK * HIDv + 255) / 256, 256, 0, stream>>>(
            gate, up, gubf, BK * HIDv);

        gemm_nt_kernel<<<dim3(Dv / 128, BK / 128), 128, 0, stream>>>(
            gubf, wd_bf, mlp, BK, Dv, HIDv, 0);

        scatter_kernel<<<BK, 256, 0, stream>>>(mlp, selidx, selpr, x);
    }

    rmsnorm_kernel<float><<<BT, 256, 0, stream>>>(
        x, final_nw, (float*)d_out, nullptr, 0);
}
